// SpanNERDecoder_63307817943769
// MI455X (gfx1250) — matrix operand order repaired and checked by reading, not compiled, not run on hardware
//
#include <hip/hip_runtime.h>

// ---------------------------------------------------------------------------
// SpanNER decoder, fused for MI455X (gfx1250, wave32).
//
//   pooled[b, s, l, :] = running-max over tokens s..s+min(l+1, S-s)-1
//   feat = [pooled | len_emb[min(l+1,S-s)-1]]  (K = 793, padded to 800)
//   logits = feat @ W + b                      (N = 9, padded to 16)
//
// Per block: 16 starts x 10 lengths = 160 spans (one batch slice).
// 10 waves; wave w owns the 16x16 output tile for span length w+1 and
// accumulates v_wmma_f32_16x16x32_bf16 over 25 K-steps.
// ---------------------------------------------------------------------------

typedef __bf16 bf16_t;
typedef __attribute__((ext_vector_type(16))) __bf16 v16bf;
typedef __attribute__((ext_vector_type(8)))  __bf16 v8bf;
typedef __attribute__((ext_vector_type(8)))  float  v8f;

namespace {
constexpr int kB = 4, kS = 512, kD = 768, kMaxLen = 10, kLenEmb = 25, kLab = 9;
constexpr int kN       = kS * kMaxLen;   // 5120 spans per batch
constexpr int kKTot    = 800;            // 768 word dims + 25 len-emb + pad
constexpr int kKSteps  = kKTot / 32;     // 25 WMMA K-steps
constexpr int kKC      = 96;             // K-chunk of word dims (3 k-steps)
constexpr int kNChunk  = kD / kKC;       // 8 word chunks (+1 len-emb tail)
constexpr int kFStride = kKC + 8;        // 104 bf16 rows -> 52 dwords: 16 tile
                                         // rows hit disjoint 4-bank groups
constexpr int kTileS   = 16;             // starts per block
constexpr int kRows    = kTileS * kMaxLen; // 160 feature rows
constexpr int kThreads = 320;            // 10 waves
} // namespace

__global__ __launch_bounds__(kThreads)
void span_ner_fused_kernel(const float* __restrict__ emb,     // (B,S,D)
                           const float* __restrict__ lenTab,  // (10,25)
                           const float* __restrict__ W,       // (793,9)
                           const float* __restrict__ bias,    // (9)
                           float* __restrict__ out)           // (B,N,9)
{
    __shared__ bf16_t Fsh[kRows * kFStride];      // pooled feature K-chunk
    __shared__ bf16_t Bsh[kKSteps * 32 * 16];     // W in B-fragment layout

    const int tid  = threadIdx.x;
    const int wave = tid >> 5;          // 0..9 -> span length wave+1
    const int lane = tid & 31;
    const int bb   = blockIdx.x >> 5;                 // batch index
    const int s0   = (blockIdx.x & 31) * kTileS;      // first start of tile

    // ---- phase 0: pack W (fp32 -> bf16) into per-lane WMMA B fragments ----
    // B (32x16 bf16): lanes 0-15 hold K=k0+{0..7,16..23} for col N=lane,
    //                 lanes 16-31 hold K=k0+{8..15,24..31} for col N=lane-16.
    for (int idx = tid; idx < kKSteps * 512; idx += kThreads) {
        const int step = idx >> 9;
        const int rem  = idx & 511;
        const int ln   = rem >> 4;
        const int i    = rem & 15;
        const int n    = ln & 15;
        const int kg   = step * 32 + ((ln >= 16) ? 8 : 0) + ((i < 8) ? i : 8 + i);
        float v = 0.0f;
        if (kg < kD + kLenEmb && n < kLab) v = W[kg * kLab + n];
        Bsh[idx] = (bf16_t)v;
    }

    v8f acc = {};   // 16x16 f32 accumulator tile for this wave

    const int rowA = wave * 16 + (lane & 15);   // feature row for A fragment
    const int hi   = lane >> 4;                 // lane half (ISA A layout)

    for (int c = 0; c <= kNChunk; ++c) {
        __syncthreads();   // WAR on Fsh vs previous chunk's reads (+ Bsh RAW)

        if (c < kNChunk) {
            // ---- pooling: running max over tokens, one (start,k) per thread
            const int k0 = c * kKC;
            for (int idx = tid; idx < kTileS * kKC; idx += kThreads) {
                const int sl = idx / kKC;
                const int k  = idx - sl * kKC;
                const int s  = s0 + sl;
                const float* p =
                    emb + ((size_t)bb * kS + s) * kD + (k0 + k);
                float run = p[0];
                Fsh[sl * kFStride + k] = (bf16_t)run;   // row t=0
                #pragma unroll
                for (int t = 1; t < kMaxLen; ++t) {
                    if (s + t < kS) run = fmaxf(run, p[t * kD]);
                    Fsh[(t * kTileS + sl) * kFStride + k] = (bf16_t)run;
                }
            }
        } else {
            // ---- tail chunk: length embedding (25 cols) + zero pad to 32
            for (int idx = tid; idx < kRows * 32; idx += kThreads) {
                const int row = idx >> 5;
                const int j   = idx & 31;
                const int t   = row >> 4;
                const int sl  = row & 15;
                int len = t + 1;
                const int rem = kS - (s0 + sl);
                if (len > rem) len = rem;
                float v = (j < kLenEmb) ? lenTab[(len - 1) * kLenEmb + j]
                                        : 0.0f;
                Fsh[row * kFStride + j] = (bf16_t)v;
            }
        }
        __syncthreads();

        // ---- WMMA over this chunk's K-steps (EXEC all-ones here) ----
        const int nsteps = (c < kNChunk) ? (kKC / 32) : 1;
        for (int kk = 0; kk < nsteps; ++kk) {
            // A (16x32 bf16): lane<16 -> K {0..7, 16..23}; lane>=16 -> +8.
            const int klb = kk * 32 + 8 * hi;          // 16B aligned
            const bf16_t* ap = &Fsh[rowA * kFStride + klb];
            const v8bf a_lo = *(const v8bf*)(ap);
            const v8bf a_hi = *(const v8bf*)(ap + 16);
            const v16bf a = __builtin_shufflevector(
                a_lo, a_hi, 0, 1, 2, 3, 4, 5, 6, 7,
                            8, 9, 10, 11, 12, 13, 14, 15);

            const int step = (c < kNChunk) ? (c * (kKC / 32) + kk)
                                           : (kKSteps - 1);
            const v16bf bf = *(const v16bf*)&Bsh[(step * 32 + lane) * 16];

            acc = __builtin_amdgcn_wmma_f32_16x16x32_bf16(
                /*neg_a=*/false, a, /*neg_b=*/false, bf,
                /*c_mod=*/(short)0, acc, /*reuse_a=*/false, /*reuse_b=*/false);
        }
    }

    // ---- bias + scatter store: D tile VGPR r holds M=r (lanes 0-15),
    //      M=r+8 (lanes 16-31); column N = lane & 15. ----
    const int n = lane & 15;
    if (n < kLab) {
        const float bn = bias[n];
        #pragma unroll
        for (int r = 0; r < 8; ++r) {
            const int sl   = r + 8 * hi;                    // start within tile
            const int span = (s0 + sl) * kMaxLen + wave;    // n index in (B,N)
            out[((size_t)bb * kN + span) * kLab + n] = acc[r] + bn;
        }
    }
}

extern "C" void kernel_launch(void* const* d_in, const int* in_sizes, int n_in,
                              void* d_out, int out_size, void* d_ws,
                              size_t ws_size, hipStream_t stream) {
    (void)in_sizes; (void)n_in; (void)d_ws; (void)ws_size; (void)out_size;
    const float* emb    = (const float*)d_in[0];  // word_embeddings (B,S,D)
    // d_in[1] span_starts / d_in[2] span_lens encode the fixed enumeration
    // (start = n/10, len = min(n%10+1, S-start)); the kernel exploits that
    // nesting to turn the gather+max into a running max.
    const float* lenTab = (const float*)d_in[3];  // len_emb_table (10,25)
    const float* W      = (const float*)d_in[4];  // (793,9)
    const float* bias   = (const float*)d_in[5];  // (9,)
    float* out          = (float*)d_out;          // (B,N,9) fp32

    const int grid = kB * (kS / kTileS);          // 4 * 32 = 128 blocks
    span_ner_fused_kernel<<<grid, kThreads, 0, stream>>>(emb, lenTab, W, bias,
                                                         out);
}
